// FCN_13855564497555
// MI455X (gfx1250) — matrix operand order repaired and checked by reading, not compiled, not run on hardware
//
#include <hip/hip_runtime.h>

typedef __attribute__((ext_vector_type(16))) __bf16 v16bf;
typedef __attribute__((ext_vector_type(8)))  float  v8f;
typedef int v4i_vs __attribute__((vector_size(16)));
typedef __attribute__((address_space(1))) v4i_vs* gptr_b128;
typedef __attribute__((address_space(3))) v4i_vs* lptr_b128;

#define B_DIM   8192
#define IN_DIM  1024
#define H_DIM   2048
#define OUT_DIM 128

#define BM 128
#define BN 128
#define BK 64
#define LDSROW (BK + 8)   // 72 bf16 elems -> 144B row stride (16B aligned, stride-36-bank: conflict free)

// ---- CDNA5 async global->LDS path (guarded: falls back to reg staging if absent) ----
#if defined(__HIP_DEVICE_COMPILE__) && __has_builtin(__builtin_amdgcn_global_load_async_to_lds_b128)
#define USE_ASYNC_LDS 1
#else
#define USE_ASYNC_LDS 0
#endif

__device__ __forceinline__ void async_copy_b128(const void* gptr, void* lptr) {
#if USE_ASYNC_LDS
  __builtin_amdgcn_global_load_async_to_lds_b128(
      (gptr_b128)gptr, (lptr_b128)lptr, 0 /*imm offset*/, 0 /*cpol*/);
#else
  (void)gptr; (void)lptr;
#endif
}

__device__ __forceinline__ void wait_async0() {
#if USE_ASYNC_LDS
#if __has_builtin(__builtin_amdgcn_s_wait_asynccnt)
  __builtin_amdgcn_s_wait_asynccnt(0);
#else
  asm volatile("s_wait_asynccnt 0x0" ::: "memory");
#endif
#endif
}

__device__ __forceinline__ __bf16 f32_to_bf16(float f) {
  union { float f; unsigned u; } in; in.f = f;
  unsigned r = in.u + 0x7FFFu + ((in.u >> 16) & 1u);  // round-to-nearest-even
  unsigned short h = (unsigned short)(r >> 16);
  union { unsigned short s; __bf16 b; } out; out.s = h;
  return out.b;
}

// vectorized fp32 -> bf16 (n must be a multiple of 8; all our sizes are)
__global__ void cvt_f32_to_bf16_v8(const float* __restrict__ in, __bf16* __restrict__ out, int n) {
  int i = (blockIdx.x * blockDim.x + threadIdx.x) * 8;
  if (i < n) {
    float4 f0 = *(const float4*)(in + i);
    float4 f1 = *(const float4*)(in + i + 4);
    alignas(16) __bf16 t[8];
    t[0] = f32_to_bf16(f0.x); t[1] = f32_to_bf16(f0.y);
    t[2] = f32_to_bf16(f0.z); t[3] = f32_to_bf16(f0.w);
    t[4] = f32_to_bf16(f1.x); t[5] = f32_to_bf16(f1.y);
    t[6] = f32_to_bf16(f1.z); t[7] = f32_to_bf16(f1.w);
    *(uint4*)(out + i) = *(const uint4*)t;
  }
}

// beta (H, OUT) fp32 -> betaT (OUT, H) bf16: readout becomes the same NT GEMM.
__global__ void transpose_beta_bf16(const float* __restrict__ beta, __bf16* __restrict__ betaT, int n) {
  int i = blockIdx.x * blockDim.x + threadIdx.x;
  if (i < n) {
    int o = i / H_DIM, k = i % H_DIM;
    betaT[(size_t)o * H_DIM + k] = f32_to_bf16(beta[(size_t)k * OUT_DIM + o]);
  }
}

// Out[m,n] = act( scale * sum_k A[m,k] * Wt[n,k] )
// A: M x K bf16 row-major.  Wt: N x K bf16 row-major (NT GEMM, both K-contiguous).
// 256 thr = 8 wave32, block tile 128x128xBK64, wave tile 64x32 (4x2 frags, 2 k-substeps -> 16 WMMA/iter).
// LDS double-buffered; next tile streamed with GLOBAL_LOAD_ASYNC_TO_LDS (ASYNCcnt), no VGPR staging.
template <bool RELU, bool OUT_BF16>
__global__ __launch_bounds__(256, 1) void gemm_nt_wmma(
    const __bf16* __restrict__ A,
    const __bf16* __restrict__ Wt,
    void* __restrict__ Out,
    int M, int N, int K, float scale)
{
  __shared__ alignas(16) __bf16 As[2][BM * LDSROW];
  __shared__ alignas(16) __bf16 Ws[2][BN * LDSROW];

  const int tid   = threadIdx.x;
  const int lane  = tid & 31;
  const int wave  = tid >> 5;
  const int waveM = wave >> 2;    // 0..1 -> 64-row band
  const int waveN = wave & 3;     // 0..3 -> 32-col band
  const int half  = lane >> 4;
  const int l     = lane & 15;

  const int m0 = blockIdx.y * BM;
  const int n0 = blockIdx.x * BN;

  // --- staging: each thread async-copies 64B of A and 64B of W per tile (4+4 x b128) ---
  const int r   = tid >> 1;          // 0..127 (row)
  const int c32 = (tid & 1) * 32;    // 0 or 32 (elem offset within 64-elem row)
  const __bf16* aPtr = A  + (size_t)(m0 + r) * K + c32;
  const __bf16* wPtr = Wt + (size_t)(n0 + r) * K + c32;

#if USE_ASYNC_LDS
  auto stage_tile = [&](int kt, int buf) {
    const __bf16* ap = aPtr + kt * BK;
    const __bf16* wp = wPtr + kt * BK;
    #pragma unroll
    for (int j = 0; j < 4; ++j) {
      async_copy_b128(ap + j * 8, &As[buf][r * LDSROW + c32 + j * 8]);
      async_copy_b128(wp + j * 8, &Ws[buf][r * LDSROW + c32 + j * 8]);
    }
  };
#else
  uint4 ra[4], rw[4];
  auto ld_tile = [&](int kt) {
    const __bf16* ap = aPtr + kt * BK;
    const __bf16* wp = wPtr + kt * BK;
    #pragma unroll
    for (int j = 0; j < 4; ++j) {
      ra[j] = *(const uint4*)(ap + j * 8);
      rw[j] = *(const uint4*)(wp + j * 8);
    }
  };
  auto st_tile = [&](int buf) {
    #pragma unroll
    for (int j = 0; j < 4; ++j) {
      *(uint4*)(&As[buf][r * LDSROW + c32 + j * 8]) = ra[j];
      *(uint4*)(&Ws[buf][r * LDSROW + c32 + j * 8]) = rw[j];
    }
  };
#endif

  const int NT = K / BK;
#if USE_ASYNC_LDS
  stage_tile(0, 0);
  wait_async0();
#else
  ld_tile(0);
  st_tile(0);
#endif
  __syncthreads();

  v8f acc[4][2] = {};
  int cur = 0;

  #pragma unroll 1
  for (int kt = 0; kt < NT; ++kt) {
#if USE_ASYNC_LDS
    // fire-and-forget: copy engine fills the other buffer while we run 16 WMMAs
    if (kt + 1 < NT) stage_tile(kt + 1, cur ^ 1);
#else
    if (kt + 1 < NT) ld_tile(kt + 1);
#endif

    #pragma unroll
    for (int ks = 0; ks < 2; ++ks) {           // two k=32 substeps of the 64-deep tile
      union Frag { v16bf v; uint4 q[2]; };
      Frag afrag[4], bfrag[2];
      #pragma unroll
      for (int mi = 0; mi < 4; ++mi) {
        // A 16x32 bf16 layout: lanes 0-15 row l hold K{0..7,16..23}; lanes 16-31 K{8..15,24..31}
        const __bf16* arow = &As[cur][(waveM * 64 + mi * 16 + l) * LDSROW + ks * 32];
        afrag[mi].q[0] = *(const uint4*)(arow + half * 8);
        afrag[mi].q[1] = *(const uint4*)(arow + half * 8 + 16);
      }
      #pragma unroll
      for (int ni = 0; ni < 2; ++ni) {
        // B 32x16 layout: lane n<16 -> col n K{0..7,16..23}; lane n+16 -> col n K{8..15,24..31}
        const __bf16* wrow = &Ws[cur][(waveN * 32 + ni * 16 + l) * LDSROW + ks * 32];
        bfrag[ni].q[0] = *(const uint4*)(wrow + half * 8);
        bfrag[ni].q[1] = *(const uint4*)(wrow + half * 8 + 16);
      }
      #pragma unroll
      for (int mi = 0; mi < 4; ++mi)
        #pragma unroll
        for (int ni = 0; ni < 2; ++ni)
          acc[mi][ni] = __builtin_amdgcn_wmma_f32_16x16x32_bf16(
              false, afrag[mi].v, false, bfrag[ni].v, (short)0, acc[mi][ni], false, false);
    }

#if USE_ASYNC_LDS
    if (kt + 1 < NT) wait_async0();     // our async writes for the next buffer landed in LDS
#else
    if (kt + 1 < NT) st_tile(cur ^ 1);
#endif

    __syncthreads();
    cur ^= 1;
  }

  // --- fused epilogue: C layout VGPR rr -> M = rr + 8*half, N = l ---
  #pragma unroll
  for (int mi = 0; mi < 4; ++mi) {
    #pragma unroll
    for (int ni = 0; ni < 2; ++ni) {
      int col     = n0 + waveN * 32 + ni * 16 + l;
      int rowbase = m0 + waveM * 64 + mi * 16 + half * 8;
      #pragma unroll
      for (int rr = 0; rr < 8; ++rr) {
        float v = acc[mi][ni][rr] * scale;
        if (RELU) v = fmaxf(v, 0.0f);
        size_t idx = (size_t)(rowbase + rr) * N + col;
        if (OUT_BF16) ((__bf16*)Out)[idx] = f32_to_bf16(v);
        else          ((float*)Out)[idx]  = v;
      }
    }
  }
}

extern "C" void kernel_launch(void* const* d_in, const int* in_sizes, int n_in,
                              void* d_out, int out_size, void* d_ws, size_t ws_size,
                              hipStream_t stream) {
  const float* x    = (const float*)d_in[0];
  const float* w0   = (const float*)d_in[1];
  const float* w1   = (const float*)d_in[2];
  const float* w2   = (const float*)d_in[3];
  const float* w3   = (const float*)d_in[4];
  const float* beta = (const float*)d_in[5];

  char* ws = (char*)d_ws;
  size_t off = 0;
  auto carve = [&](size_t bytes) -> void* {
    void* p = ws + off;
    off += (bytes + 255) & ~(size_t)255;
    return p;
  };
  __bf16* x_bf  = (__bf16*)carve((size_t)B_DIM * IN_DIM * 2);    // 16 MB
  __bf16* w0_bf = (__bf16*)carve((size_t)H_DIM * IN_DIM * 2);    //  4 MB
  __bf16* w1_bf = (__bf16*)carve((size_t)H_DIM * H_DIM * 2);     //  8 MB
  __bf16* w2_bf = (__bf16*)carve((size_t)H_DIM * H_DIM * 2);     //  8 MB
  __bf16* w3_bf = (__bf16*)carve((size_t)H_DIM * H_DIM * 2);     //  8 MB
  __bf16* bT_bf = (__bf16*)carve((size_t)OUT_DIM * H_DIM * 2);   // 0.5 MB
  __bf16* act0  = (__bf16*)carve((size_t)B_DIM * H_DIM * 2);     // 32 MB
  __bf16* act1  = (__bf16*)carve((size_t)B_DIM * H_DIM * 2);     // 32 MB

  auto cvt = [&](const float* src, __bf16* dst, size_t n) {
    int blocks = (int)((n / 8 + 255) / 256);
    cvt_f32_to_bf16_v8<<<blocks, 256, 0, stream>>>(src, dst, (int)n);
  };
  cvt(x,  x_bf,  (size_t)B_DIM * IN_DIM);
  cvt(w0, w0_bf, (size_t)H_DIM * IN_DIM);
  cvt(w1, w1_bf, (size_t)H_DIM * H_DIM);
  cvt(w2, w2_bf, (size_t)H_DIM * H_DIM);
  cvt(w3, w3_bf, (size_t)H_DIM * H_DIM);
  { int n = OUT_DIM * H_DIM; transpose_beta_bf16<<<(n + 255) / 256, 256, 0, stream>>>(beta, bT_bf, n); }

  const float s0 = 0.03125f;               // 1/sqrt(1024)
  const float s1 = 0.022097086912079612f;  // 1/sqrt(2048)
  dim3 blk(256);

  gemm_nt_wmma<true,  true ><<<dim3(H_DIM / BN, B_DIM / BM), blk, 0, stream>>>(x_bf, w0_bf, act0, B_DIM, H_DIM, IN_DIM, s0);
  gemm_nt_wmma<true,  true ><<<dim3(H_DIM / BN, B_DIM / BM), blk, 0, stream>>>(act0, w1_bf, act1, B_DIM, H_DIM, H_DIM, s1);
  gemm_nt_wmma<true,  true ><<<dim3(H_DIM / BN, B_DIM / BM), blk, 0, stream>>>(act1, w2_bf, act0, B_DIM, H_DIM, H_DIM, s1);
  gemm_nt_wmma<true,  true ><<<dim3(H_DIM / BN, B_DIM / BM), blk, 0, stream>>>(act0, w3_bf, act1, B_DIM, H_DIM, H_DIM, s1);
  gemm_nt_wmma<false, false><<<dim3(OUT_DIM / BN, B_DIM / BM), blk, 0, stream>>>(act1, bT_bf, d_out, B_DIM, OUT_DIM, H_DIM, 1.0f / (float)H_DIM);
}
